// MultiHeadAttention_GPT_2_39230231281982
// MI455X (gfx1250) — matrix operand order repaired and checked by reading, not compiled, not run on hardware
//
#include <hip/hip_runtime.h>
#include <hip/hip_bf16.h>

// GPT-2 MHA forward on gfx1250 (MI455X): f16 WMMA everywhere, f32 accumulate.
// GEMMs use the Tensor Data Mover (tensor_load_to_lds) with double-buffered
// LDS; attention is register-resident flash attention.
// B=4, T=1024, C=1024, H=16, DH=64.

typedef _Float16 h16;
typedef __attribute__((ext_vector_type(16))) _Float16 v16h;
typedef __attribute__((ext_vector_type(8)))  float    v8f;
typedef __attribute__((ext_vector_type(4)))  float    v4f;
typedef __attribute__((ext_vector_type(4)))  unsigned int v4u;
typedef __attribute__((ext_vector_type(8)))  int      v8i;
typedef __attribute__((ext_vector_type(4)))  int      v4i;

#define TT  1024
#define CCH 1024
#define HH  16
#define DHH 64
#define MM  4096   // B*T

union Frag {
  v16h v;
  v4f  f4[2];
};

// A-matrix fragment (16x32 f16, MxK). Lane m (0-15): K0-7 then K16-23.
// Lane m+16: K8-15 then K24-31.  base = &mat[row0][k0], row-major, ld in halfs.
__device__ __forceinline__ v16h load_a_frag(const h16* base, int ld, int lane) {
  Frag f;
  const h16* p = base + (size_t)(lane & 15) * ld + ((lane >> 4) << 3);
  f.f4[0] = *(const v4f*)(p);
  f.f4[1] = *(const v4f*)(p + 16);
  return f.v;
}

// B-matrix fragment (32x16 f16, KxN), column-major source (i.e. Bt row-major,
// row index = N column). Lane n (0-15): K0-7 then K16-23; lane n+16: K8-15 then K24-31.
__device__ __forceinline__ v16h load_b_frag(const h16* base, int ld, int lane) {
  Frag f;
  const h16* p = base + (size_t)(lane & 15) * ld + ((lane >> 4) << 3);
  f.f4[0] = *(const v4f*)(p);
  f.f4[1] = *(const v4f*)(p + 16);
  return f.v;
}

__device__ __forceinline__ v8f wmma16(v16h a, v16h b, v8f c) {
  return __builtin_amdgcn_wmma_f32_16x16x32_f16(false, a, false, b, (short)0, c,
                                                false, false);
}

// ---- Tensor Data Mover: DMA a 2D f16 tile (32 cols x tile_rows) from a
// row-major tensor (ld = 1024 halfs) into LDS, with 16B padding after every
// 64B row (LDS row stride 40 halfs = 80B). D# built per cdna5 ISA ch.8.
// Toolchain here is the 6-arg (clang-23 / therock) builtin form.
__device__ __forceinline__ void tdm_load_tile(unsigned lds_byte,
                                              const h16* gptr,
                                              unsigned tile_rows) {
  unsigned long long ga = (unsigned long long)(uintptr_t)gptr;
  v4u g0;
  g0[0] = 1u;                                   // count=1, user mode
  g0[1] = lds_byte;                             // lds_addr
  g0[2] = (unsigned)ga;                         // global_addr[31:0]
  g0[3] = (unsigned)(ga >> 32) | (2u << 30);    // global_addr[56:32] | type=2
  v8i g1;
  // data_size=1(2B), pad_enable, pad_interval=3(16 DW), pad_amount=3(4 DW)
  g1[0] = (int)((1u << 16) | (1u << 20) | (3u << 22) | (3u << 25));
  g1[1] = (int)(1024u << 16);   // tensor_dim0 = 1024 (bits 79:48)
  g1[2] = (int)(4096u << 16);   // tensor_dim1 = 4096 (bits 111:80)
  g1[3] = (int)(32u << 16);     // tile_dim0 = 32   (bits 127:112)
  g1[4] = (int)tile_rows;       // tile_dim1        (bits 143:128)
  g1[5] = 1024;                 // tensor_dim0_stride = 1024 (bits 207:160)
  g1[6] = 0;
  g1[7] = 0;
  v4i z4 = {0, 0, 0, 0};
  v8i z8 = {0, 0, 0, 0, 0, 0, 0, 0};
  __builtin_amdgcn_tensor_load_to_lds(g0, g1, z4, z4, z8, 0);
}

// ---------------- elementwise converts ----------------

__global__ void k_cvt_f16(const float* __restrict__ in, h16* __restrict__ out) {
  size_t i = (size_t)blockIdx.x * blockDim.x + threadIdx.x;
  out[i] = (h16)in[i];
}

// Wt[cout][cin] = (f16) W[cin][cout]   (W is CxC row-major)
__global__ void k_transpose_w(const float* __restrict__ W, h16* __restrict__ Wt) {
  size_t i = (size_t)blockIdx.x * blockDim.x + threadIdx.x;
  int co = (int)(i >> 10);
  int ci = (int)(i & 1023);
  Wt[i] = (h16)W[(size_t)ci * CCH + co];
}

// ---------------- WMMA GEMM: out = A[M,K] @ Wt[N,K]^T + bias ----------------
// Block tile 64(M) x 128(N), 8 waves in 2x4, each wave 32x32 (2x2 wmma tiles).
// Tiles DMA'd global->LDS by the TDM (wave 0 issues), double buffered.
// modes: 0/1 -> f16 heads layout [B,H,T,DH]; 2 -> f16 [B,H,DH,T]; 3 -> f32 flat.

__global__ __launch_bounds__(256)
void k_gemm(const h16* __restrict__ A, const h16* __restrict__ Bt,
            const float* __restrict__ bias, void* __restrict__ out, int mode) {
  __shared__ h16 sA[2][64 * 40];
  __shared__ h16 sB[2][128 * 40];

  const int tid  = threadIdx.x;
  const int lane = tid & 31;
  const int w    = tid >> 5;
  const int wm   = w >> 2;   // 0..1
  const int wn   = w & 3;    // 0..3
  const int m0   = blockIdx.y * 64;
  const int n0   = blockIdx.x * 128;

  const unsigned ldsA[2] = {(unsigned)(uintptr_t)&sA[0][0],
                            (unsigned)(uintptr_t)&sA[1][0]};
  const unsigned ldsB[2] = {(unsigned)(uintptr_t)&sB[0][0],
                            (unsigned)(uintptr_t)&sB[1][0]};
  const h16* gA = A + (size_t)m0 * CCH;
  const h16* gB = Bt + (size_t)n0 * CCH;

  v8f acc[2][2] = {};

  if (w == 0) {
    tdm_load_tile(ldsA[0], gA, 64);
    tdm_load_tile(ldsB[0], gB, 128);
  }

  const int nsteps = CCH / 32;
  for (int i = 0; i < nsteps; ++i) {
    if (w == 0) {
      if (i + 1 < nsteps) {
        const int kn = (i + 1) * 32;
        tdm_load_tile(ldsA[(i + 1) & 1], gA + kn, 64);
        tdm_load_tile(ldsB[(i + 1) & 1], gB + kn, 128);
        __builtin_amdgcn_s_wait_tensorcnt(2);  // current buffer's pair done
      } else {
        __builtin_amdgcn_s_wait_tensorcnt(0);
      }
    }
    __syncthreads();

    const h16* cA = &sA[i & 1][0];
    const h16* cB = &sB[i & 1][0];
    v16h aF[2], bF[2];
#pragma unroll
    for (int a = 0; a < 2; ++a)
      aF[a] = load_a_frag(cA + (wm * 32 + a * 16) * 40, 40, lane);
#pragma unroll
    for (int b = 0; b < 2; ++b)
      bF[b] = load_b_frag(cB + (wn * 32 + b * 16) * 40, 40, lane);

#pragma unroll
    for (int a = 0; a < 2; ++a)
#pragma unroll
      for (int b = 0; b < 2; ++b)
        acc[a][b] = wmma16(aF[a], bF[b], acc[a][b]);
    __syncthreads();
  }

  const int l15 = lane & 15;
  const int hi8 = (lane & 16) >> 1;  // 0 or 8

#pragma unroll
  for (int i = 0; i < 2; ++i) {
#pragma unroll
    for (int j = 0; j < 2; ++j) {
      const int colb = n0 + wn * 32 + j * 16 + l15;
      const float bv = bias[colb];
#pragma unroll
      for (int jv = 0; jv < 8; ++jv) {
        const int row = m0 + wm * 32 + i * 16 + jv + hi8;
        const float val = acc[i][j][jv] + bv;
        if (mode == 3) {
          ((float*)out)[(size_t)row * CCH + colb] = val;
        } else {
          const int b  = row >> 10, t = row & 1023;
          const int hh = colb >> 6, dh = colb & 63;
          if (mode == 2)  // V transposed: [B,H,DH,T]
            ((h16*)out)[(((size_t)b * HH + hh) * DHH + dh) * TT + t] = (h16)val;
          else            // Q/K heads: [B,H,T,DH]
            ((h16*)out)[(((size_t)b * HH + hh) * TT + t) * DHH + dh] = (h16)val;
        }
      }
    }
  }
}

// ---------------- flash attention (causal), one wave per 16-query tile ------
// S^T = K . Q^T via WMMA -> softmaxed S^T lanes are exactly the A-fragment of
// P, so P is built fully in-lane. O += P . V with V stored [DH][T].

__global__ __launch_bounds__(128)
void k_attn(const h16* __restrict__ Qh, const h16* __restrict__ Kh,
            const h16* __restrict__ Vt, h16* __restrict__ ctxh) {
  const int lane = threadIdx.x & 31;
  const int wave = threadIdx.x >> 5;  // 0..3
  const int bh   = blockIdx.y;        // 0..63
  const int b    = bh >> 4;
  const int h    = bh & 15;
  const int qb   = (blockIdx.x * 4 + wave) * 16;

  const h16* Q = Qh + (size_t)bh * TT * DHH;
  const h16* K = Kh + (size_t)bh * TT * DHH;
  const h16* V = Vt + (size_t)bh * DHH * TT;

  // Q tile as B-matrix of S^T = K.Q^T  (two K-chunks over DH=64)
  const v16h qf0 = load_b_frag(Q + (size_t)qb * DHH + 0, DHH, lane);
  const v16h qf1 = load_b_frag(Q + (size_t)qb * DHH + 32, DHH, lane);

  v8f o[4] = {};
  float m_run = -1e30f, l_run = 0.0f;
  const int l15 = lane & 15;
  const int hi8 = (lane & 16) >> 1;   // 0 or 8
  const int q_lane = qb + l15;

  for (int kb = 0; kb < qb + 16; kb += 32) {
    // prefetch next kv tile while computing this one
    if (kb + 32 < qb + 16) {
      __builtin_prefetch(K + (size_t)(kb + 32) * DHH + lane * 16, 0, 1);
      __builtin_prefetch(V + (size_t)lane * TT + kb + 32, 0, 1);
    }

    // S^T tiles: A = K rows (keys), B = Q^T
    v8f s0 = {}, s1 = {};
    s0 = wmma16(load_a_frag(K + (size_t)kb * DHH + 0, DHH, lane), qf0, s0);
    s0 = wmma16(load_a_frag(K + (size_t)kb * DHH + 32, DHH, lane), qf1, s0);
    s1 = wmma16(load_a_frag(K + (size_t)(kb + 16) * DHH + 0, DHH, lane), qf0, s1);
    s1 = wmma16(load_a_frag(K + (size_t)(kb + 16) * DHH + 32, DHH, lane), qf1, s1);

    // scale + causal mask.  s* layout: vgpr jv -> key = kb(+16) + jv + hi8,
    // q = q_lane (column = lane&15).
    float e0[8], e1[8];
    float mloc = -1e30f;
#pragma unroll
    for (int jv = 0; jv < 8; ++jv) {
      float a = s0[jv] * 0.125f;                 // DH^-0.5
      float c = s1[jv] * 0.125f;
      if (kb + jv + hi8 > q_lane)       a = -1e30f;
      if (kb + 16 + jv + hi8 > q_lane)  c = -1e30f;
      e0[jv] = a; e1[jv] = c;
      mloc = fmaxf(mloc, fmaxf(a, c));
    }
    mloc = fmaxf(mloc, __shfl_xor(mloc, 16, 32));
    const float m_new = fmaxf(m_run, mloc);
    const float alpha = __expf(m_run - m_new);

    float sloc = 0.0f;
#pragma unroll
    for (int jv = 0; jv < 8; ++jv) {
      e0[jv] = __expf(e0[jv] - m_new);
      e1[jv] = __expf(e1[jv] - m_new);
      sloc += e0[jv] + e1[jv];
    }
    sloc += __shfl_xor(sloc, 16, 32);
    l_run = l_run * alpha + sloc;
    m_run = m_new;

    // rescale O: row q of accumulator vgpr jv is jv+hi8 -> broadcast alpha
#pragma unroll
    for (int jv = 0; jv < 8; ++jv) {
      const float aj = __shfl(alpha, jv + hi8, 32);
      o[0][jv] *= aj; o[1][jv] *= aj; o[2][jv] *= aj; o[3][jv] *= aj;
    }

    // P as A-fragment: elements 0-7 = keys (0-7|8-15), 8-15 = keys (16-23|24-31)
    Frag pf;
#pragma unroll
    for (int jv = 0; jv < 8; ++jv) {
      pf.v[jv]     = (h16)e0[jv];
      pf.v[8 + jv] = (h16)e1[jv];
    }

    // O += P . V  over 4 N-tiles of DH
#pragma unroll
    for (int n = 0; n < 4; ++n) {
      v16h vf = load_b_frag(V + (size_t)(n * 16) * TT + kb, TT, lane);
      o[n] = wmma16(pf.v, vf, o[n]);
    }
  }

  // epilogue: divide by row sums, store ctx f16 [B*T, C]
#pragma unroll
  for (int jv = 0; jv < 8; ++jv) {
    const float lj = __shfl(l_run, jv + hi8, 32);
    const float rj = 1.0f / lj;
    const int row = qb + jv + hi8;
#pragma unroll
    for (int n = 0; n < 4; ++n) {
      const int col = h * DHH + n * 16 + l15;
      ctxh[((size_t)b * TT + row) * CCH + col] = (h16)(o[n][jv] * rj);
    }
  }
}

// ---------------------------------------------------------------------------

extern "C" void kernel_launch(void* const* d_in, const int* in_sizes, int n_in,
                              void* d_out, int out_size, void* d_ws, size_t ws_size,
                              hipStream_t stream) {
  (void)in_sizes; (void)n_in; (void)out_size; (void)ws_size;

  const float* x  = (const float*)d_in[0];
  const float* Wq = (const float*)d_in[1];
  const float* bq = (const float*)d_in[2];
  const float* Wk = (const float*)d_in[3];
  const float* bk = (const float*)d_in[4];
  const float* Wv = (const float*)d_in[5];
  const float* bv = (const float*)d_in[6];
  const float* Wp = (const float*)d_in[7];
  const float* bp = (const float*)d_in[8];

  char* ws = (char*)d_ws;
  h16* xh   = (h16*)(ws);                          // 8 MB
  h16* wtq  = (h16*)(ws + ((size_t)8  << 20));     // 2 MB each
  h16* wtk  = (h16*)(ws + ((size_t)10 << 20));
  h16* wtv  = (h16*)(ws + ((size_t)12 << 20));
  h16* wtp  = (h16*)(ws + ((size_t)14 << 20));
  h16* Qh   = (h16*)(ws + ((size_t)16 << 20));     // 8 MB
  h16* Kh   = (h16*)(ws + ((size_t)24 << 20));     // 8 MB
  h16* Vt   = (h16*)(ws + ((size_t)32 << 20));     // 8 MB
  h16* ctxh = (h16*)(ws + ((size_t)40 << 20));     // 8 MB

  // converts
  k_cvt_f16<<<(MM * CCH) / 256, 256, 0, stream>>>(x, xh);
  k_transpose_w<<<(CCH * CCH) / 256, 256, 0, stream>>>(Wq, wtq);
  k_transpose_w<<<(CCH * CCH) / 256, 256, 0, stream>>>(Wk, wtk);
  k_transpose_w<<<(CCH * CCH) / 256, 256, 0, stream>>>(Wv, wtv);
  k_transpose_w<<<(CCH * CCH) / 256, 256, 0, stream>>>(Wp, wtp);

  // QKV projections
  dim3 ggrid(CCH / 128, MM / 64);
  k_gemm<<<ggrid, 256, 0, stream>>>(xh, wtq, bq, (void*)Qh, 0);
  k_gemm<<<ggrid, 256, 0, stream>>>(xh, wtk, bk, (void*)Kh, 1);
  k_gemm<<<ggrid, 256, 0, stream>>>(xh, wtv, bv, (void*)Vt, 2);

  // causal flash attention
  k_attn<<<dim3(TT / 64, 4 * HH), 128, 0, stream>>>(Qh, Kh, Vt, ctxh);

  // output projection (f32 + bias) into d_out
  k_gemm<<<ggrid, 256, 0, stream>>>(ctxh, wtp, bp, d_out, 3);
}
